// GATv2Net_54168127537215
// MI455X (gfx1250) — compile-verified, matrix operand order checked by reading, not run on hardware
//
#include <hip/hip_runtime.h>
#include <hip/hip_bf16.h>

typedef __attribute__((ext_vector_type(16))) _Float16 v16h;
typedef __attribute__((ext_vector_type(8)))  float    v8f;

#define NNODES 50000
#define INCH   128
#define HIDC   64
#define HEADS  4
#define OUTC   16
#define EORIG  800000
#define ETOT   (EORIG + NNODES)
#define F1     (HEADS * HIDC)   // 256
#define NEG_SLOPE 0.2f

// ---- order-preserving float <-> uint key for atomicMax on signed floats ----
__device__ __forceinline__ unsigned fkey(float f) {
  unsigned u = __float_as_uint(f);
  return (u & 0x80000000u) ? ~u : (u | 0x80000000u);
}
__device__ __forceinline__ float fdecode(unsigned k) {
  unsigned u = (k & 0x80000000u) ? (k & 0x7FFFFFFFu) : ~k;
  return __uint_as_float(u);
}
__device__ __forceinline__ float lrelu(float v) { return v > 0.f ? v : NEG_SLOPE * v; }

// =====================================================================
// WMMA GEMM: C[M,Nc] = A[M,K] * W[K,Nc]   (f32 memory, f16 math, f32 acc)
// Compile-time K/Nc/NT. One wave computes NT adjacent 16x16 tiles in N,
// reusing a single A fragment across NT WMMAs per K-step (4x less A traffic,
// 4 independent accumulator chains -> no WMMA->WMMA RAW stalls).
// M must be a multiple of 16 (50000 = 3125*16).
// A-fragment layout (16-bit A 16x32, ISA 7.12.2): lane L holds row L%16,
// elems 0..7 -> K = 8*(L/16)+0..7, elems 8..15 -> K = 16+8*(L/16)+0..7.
// B symmetric (col = L%16, same K mapping). C/D: lane->col L%16, VGPR v ->
// row v + 8*(L/16).
// =====================================================================
template <int K, int Nc, int NT>
__global__ void k_wmma_gemm(const float* __restrict__ A, const float* __restrict__ W,
                            float* __restrict__ C, int M) {
  constexpr int GROUPS_N = Nc / (16 * NT);
  int wave = (blockIdx.x * blockDim.x + threadIdx.x) >> 5;
  int lane = threadIdx.x & 31;
  int tileM = wave / GROUPS_N;
  int g     = wave - tileM * GROUPS_N;
  if (tileM * 16 >= M) return;              // wave-uniform: EXEC stays all-1s
  int r    = lane & 15;
  int hf   = lane >> 4;
  int row  = tileM * 16 + r;
  int col0 = g * (16 * NT) + r;

  v8f acc[NT];
#pragma unroll
  for (int t = 0; t < NT; ++t) acc[t] = (v8f){};

  const float* arow = A + (size_t)row * K;
#pragma unroll
  for (int kk = 0; kk < K; kk += 32) {
    v16h a;
    const float* ap = arow + kk + 8 * hf;
#pragma unroll
    for (int j = 0; j < 8; ++j) {
      a[j]     = (_Float16)ap[j];
      a[j + 8] = (_Float16)ap[j + 16];
    }
#pragma unroll
    for (int t = 0; t < NT; ++t) {
      v16h b;
      const float* wp = W + (size_t)(kk + 8 * hf) * Nc + col0 + t * 16;
#pragma unroll
      for (int j = 0; j < 8; ++j) {
        b[j]     = (_Float16)wp[(size_t)j * Nc];
        b[j + 8] = (_Float16)wp[(size_t)(j + 16) * Nc];
      }
      acc[t] = __builtin_amdgcn_wmma_f32_16x16x32_f16(false, a, false, b,
                                                      (short)0, acc[t], false, false);
    }
  }
#pragma unroll
  for (int t = 0; t < NT; ++t) {
    float* cp = C + (size_t)(tileM * 16 + 8 * hf) * Nc + col0 + t * 16;
#pragma unroll
    for (int v = 0; v < 8; ++v) cp[(size_t)v * Nc] = acc[t][v];
  }
}

// =====================================================================
// Init kernels (accumulators seeded with bias; max keys to -inf-ish; sums 0)
// =====================================================================
__global__ void k_init1(float* __restrict__ acc1, const float* __restrict__ b1,
                        unsigned* __restrict__ maxk1, float* __restrict__ sum1) {
  int i = blockIdx.x * blockDim.x + threadIdx.x;
  if (i < NNODES * F1) acc1[i] = b1[i & (F1 - 1)];
  if (i < NNODES * HEADS) { maxk1[i] = fkey(-3.0e38f); sum1[i] = 0.f; }
}
__global__ void k_init2(float* __restrict__ out, const float* __restrict__ b2,
                        unsigned* __restrict__ maxk2, float* __restrict__ sum2) {
  int i = blockIdx.x * blockDim.x + threadIdx.x;
  if (i < NNODES * OUTC) out[i] = b2[i & (OUTC - 1)];
  if (i < NNODES) { maxk2[i] = fkey(-3.0e38f); sum2[i] = 0.f; }
}

// =====================================================================
// Layer-1 edge kernels (wave per edge; lane owns 8 contiguous channels)
// =====================================================================
__global__ void k_edge_logits1(const float* __restrict__ xl, const float* __restrict__ xr,
                               const long long* __restrict__ ei, const float* __restrict__ att,
                               float* __restrict__ logits, unsigned* __restrict__ maxk) {
  int wave = (blockIdx.x * blockDim.x + threadIdx.x) >> 5;
  int lane = threadIdx.x & 31;
  if (wave >= ETOT) return;
  int s, d;
  if (wave < EORIG) { s = (int)ei[wave]; d = (int)ei[EORIG + wave]; }
  else              { s = wave - EORIG; d = s; }

  const float4* pl = (const float4*)(xl + (size_t)s * F1) + lane * 2;
  const float4* pr = (const float4*)(xr + (size_t)d * F1) + lane * 2;
  const float4* pa = (const float4*)att + lane * 2;
  float acc = 0.f;
#pragma unroll
  for (int j = 0; j < 2; ++j) {
    float4 a = pl[j], b = pr[j], w = pa[j];
    acc += lrelu(a.x + b.x) * w.x;
    acc += lrelu(a.y + b.y) * w.y;
    acc += lrelu(a.z + b.z) * w.z;
    acc += lrelu(a.w + b.w) * w.w;
  }
  // reduce the 8 lanes of each head group
  acc += __shfl_xor(acc, 1, 32);
  acc += __shfl_xor(acc, 2, 32);
  acc += __shfl_xor(acc, 4, 32);
  int head = lane >> 3;
  if ((lane & 7) == 0) {
    logits[(size_t)wave * HEADS + head] = acc;
    atomicMax(&maxk[d * HEADS + head], fkey(acc));
  }
}

__global__ void k_edge_exp1(const long long* __restrict__ ei, float* __restrict__ logits,
                            const unsigned* __restrict__ maxk, float* __restrict__ sum) {
  int e = blockIdx.x * blockDim.x + threadIdx.x;
  if (e >= ETOT) return;
  int d = (e < EORIG) ? (int)ei[EORIG + e] : (e - EORIG);
#pragma unroll
  for (int h = 0; h < HEADS; ++h) {
    float m  = fdecode(maxk[d * HEADS + h]);
    float ex = __expf(logits[(size_t)e * HEADS + h] - m);
    logits[(size_t)e * HEADS + h] = ex;
    atomicAdd(&sum[d * HEADS + h], ex);
  }
}

__global__ void k_edge_scatter1(const float* __restrict__ xl, const long long* __restrict__ ei,
                                const float* __restrict__ exv, const float* __restrict__ sum,
                                float* __restrict__ acc) {
  int wave = (blockIdx.x * blockDim.x + threadIdx.x) >> 5;
  int lane = threadIdx.x & 31;
  if (wave >= ETOT) return;
  int s, d;
  if (wave < EORIG) { s = (int)ei[wave]; d = (int)ei[EORIG + wave]; }
  else              { s = wave - EORIG; d = s; }
  int head = lane >> 3;
  float alpha = exv[(size_t)wave * HEADS + head] / sum[d * HEADS + head];
  const float* pl = xl + (size_t)s * F1 + lane * 8;
  float* pd       = acc + (size_t)d * F1 + lane * 8;
#pragma unroll
  for (int j = 0; j < 8; ++j) atomicAdd(&pd[j], alpha * pl[j]);
}

__global__ void k_elu_inplace(float* __restrict__ h) {
  int i = blockIdx.x * blockDim.x + threadIdx.x;
  if (i >= NNODES * F1) return;
  float v = h[i];
  h[i] = v > 0.f ? v : (__expf(v) - 1.f);
}

// =====================================================================
// Layer-2 edge kernels (1 head, 16 channels)
// =====================================================================
__global__ void k_edge_logits2(const float* __restrict__ xl, const float* __restrict__ xr,
                               const long long* __restrict__ ei, const float* __restrict__ att,
                               float* __restrict__ logits, unsigned* __restrict__ maxk) {
  int e = blockIdx.x * blockDim.x + threadIdx.x;
  if (e >= ETOT) return;
  int s, d;
  if (e < EORIG) { s = (int)ei[e]; d = (int)ei[EORIG + e]; }
  else           { s = e - EORIG; d = s; }
  const float4* pl = (const float4*)(xl + (size_t)s * OUTC);
  const float4* pr = (const float4*)(xr + (size_t)d * OUTC);
  const float4* pa = (const float4*)att;
  float acc = 0.f;
#pragma unroll
  for (int j = 0; j < 4; ++j) {
    float4 a = pl[j], b = pr[j], w = pa[j];
    acc += lrelu(a.x + b.x) * w.x;
    acc += lrelu(a.y + b.y) * w.y;
    acc += lrelu(a.z + b.z) * w.z;
    acc += lrelu(a.w + b.w) * w.w;
  }
  logits[e] = acc;
  atomicMax(&maxk[d], fkey(acc));
}

__global__ void k_edge_exp2(const long long* __restrict__ ei, float* __restrict__ logits,
                            const unsigned* __restrict__ maxk, float* __restrict__ sum) {
  int e = blockIdx.x * blockDim.x + threadIdx.x;
  if (e >= ETOT) return;
  int d = (e < EORIG) ? (int)ei[EORIG + e] : (e - EORIG);
  float ex = __expf(logits[e] - fdecode(maxk[d]));
  logits[e] = ex;
  atomicAdd(&sum[d], ex);
}

__global__ void k_edge_scatter2(const float* __restrict__ xl, const long long* __restrict__ ei,
                                const float* __restrict__ exv, const float* __restrict__ sum,
                                float* __restrict__ out) {
  int tid = blockIdx.x * blockDim.x + threadIdx.x;
  if (tid >= ETOT * OUTC) return;
  int e = tid >> 4, c = tid & 15;
  int s, d;
  if (e < EORIG) { s = (int)ei[e]; d = (int)ei[EORIG + e]; }
  else           { s = e - EORIG; d = s; }
  float alpha = exv[e] / sum[d];
  atomicAdd(&out[(size_t)d * OUTC + c], alpha * xl[(size_t)s * OUTC + c]);
}

// =====================================================================
extern "C" void kernel_launch(void* const* d_in, const int* in_sizes, int n_in,
                              void* d_out, int out_size, void* d_ws, size_t ws_size,
                              hipStream_t stream) {
  const float*     x    = (const float*)d_in[0];
  const long long* ei   = (const long long*)d_in[1];
  const float*     W1l  = (const float*)d_in[2];
  const float*     W1r  = (const float*)d_in[3];
  const float*     att1 = (const float*)d_in[4];
  const float*     b1   = (const float*)d_in[5];
  const float*     W2l  = (const float*)d_in[6];
  const float*     W2r  = (const float*)d_in[7];
  const float*     att2 = (const float*)d_in[8];
  const float*     b2   = (const float*)d_in[9];
  float*           out  = (float*)d_out;

  // workspace carve-up
  float* p = (float*)d_ws;
  float*    xl1   = p;                 p += (size_t)NNODES * F1;
  float*    xr1   = p;                 p += (size_t)NNODES * F1;
  float*    acc1  = p;                 p += (size_t)NNODES * F1;   // becomes h in place
  float*    xl2   = p;                 p += (size_t)NNODES * OUTC;
  float*    xr2   = p;                 p += (size_t)NNODES * OUTC;
  unsigned* maxk1 = (unsigned*)p;      p += (size_t)NNODES * HEADS;
  float*    sum1  = p;                 p += (size_t)NNODES * HEADS;
  unsigned* maxk2 = (unsigned*)p;      p += (size_t)NNODES;
  float*    sum2  = p;                 p += (size_t)NNODES;
  float*    ex1   = p;                 p += (size_t)ETOT * HEADS;
  float*    ex2   = p;                 /* p += ETOT; */

  const int TB = 256;
  auto blk = [](long long n, int tb) { return (int)((n + tb - 1) / tb); };

  // init accumulators / softmax state (before any edge kernel)
  hipLaunchKernelGGL(k_init1, dim3(blk((long long)NNODES * F1, TB)), dim3(TB), 0, stream,
                     acc1, b1, maxk1, sum1);
  hipLaunchKernelGGL(k_init2, dim3(blk((long long)NNODES * OUTC, TB)), dim3(TB), 0, stream,
                     out, b2, maxk2, sum2);

  // ---- layer 1 GEMMs (WMMA): [50000,128] x [128,256], 16x64 per wave ----
  {
    long long waves = (long long)(NNODES / 16) * (F1 / 64);     // 3125 * 4 = 12500
    int blocks = blk(waves * 32, TB);
    hipLaunchKernelGGL((k_wmma_gemm<INCH, F1, 4>), dim3(blocks), dim3(TB), 0, stream,
                       x, W1l, xl1, NNODES);
    hipLaunchKernelGGL((k_wmma_gemm<INCH, F1, 4>), dim3(blocks), dim3(TB), 0, stream,
                       x, W1r, xr1, NNODES);
  }

  // ---- layer 1 attention ----
  int ew_blocks = blk((long long)ETOT * 32, TB);   // wave per edge
  int e_blocks  = blk((long long)ETOT, TB);        // thread per edge
  hipLaunchKernelGGL(k_edge_logits1, dim3(ew_blocks), dim3(TB), 0, stream,
                     xl1, xr1, ei, att1, ex1, maxk1);
  hipLaunchKernelGGL(k_edge_exp1, dim3(e_blocks), dim3(TB), 0, stream,
                     ei, ex1, maxk1, sum1);
  hipLaunchKernelGGL(k_edge_scatter1, dim3(ew_blocks), dim3(TB), 0, stream,
                     xl1, ei, ex1, sum1, acc1);
  hipLaunchKernelGGL(k_elu_inplace, dim3(blk((long long)NNODES * F1, TB)), dim3(TB), 0, stream,
                     acc1);

  // ---- layer 2 GEMMs (WMMA): [50000,256] x [256,16] ----
  {
    long long waves = (long long)(NNODES / 16);                 // 3125 waves, 1 N-tile
    int blocks = blk(waves * 32, TB);
    hipLaunchKernelGGL((k_wmma_gemm<F1, OUTC, 1>), dim3(blocks), dim3(TB), 0, stream,
                       acc1, W2l, xl2, NNODES);
    hipLaunchKernelGGL((k_wmma_gemm<F1, OUTC, 1>), dim3(blocks), dim3(TB), 0, stream,
                       acc1, W2r, xr2, NNODES);
  }

  // ---- layer 2 attention (accumulates into out, pre-seeded with b2) ----
  hipLaunchKernelGGL(k_edge_logits2, dim3(e_blocks), dim3(TB), 0, stream,
                     xl2, xr2, ei, att2, ex2, maxk2);
  hipLaunchKernelGGL(k_edge_exp2, dim3(e_blocks), dim3(TB), 0, stream,
                     ei, ex2, maxk2, sum2);
  hipLaunchKernelGGL(k_edge_scatter2, dim3(blk((long long)ETOT * OUTC, TB)), dim3(TB), 0, stream,
                     xl2, ei, ex2, sum2, out);
}
